// CrystalGraphConvolution_48627619726060
// MI455X (gfx1250) — matrix operand order repaired and checked by reading, not compiled, not run on hardware
//
#include <hip/hip_runtime.h>
#include <hip/hip_bf16.h>
#include <math.h>

// ---------- CDNA5 WMMA types ----------
typedef __attribute__((ext_vector_type(16))) _Float16 v16h;
typedef __attribute__((ext_vector_type(8)))  float    v8f;

union Frag16 { v16h v; uint4 u[2]; };
struct H4 { _Float16 a, b, c, d; };   // 8-byte packed f16 quad

// ---------- gfx1250 async-to-LDS path (guarded; falls back to VGPR copy) ----
#if __has_builtin(__builtin_amdgcn_global_load_async_to_lds_b128)
#define HAS_ASYNC_LDS 1
// builtin signature (from hipcc diagnostic): params are pointers to
// 'int __attribute__((vector_size(16)))' in addrspace(1)/(3).
typedef int v4i_ __attribute__((vector_size(16)));
#define GAS1(p) ((__attribute__((address_space(1))) v4i_*)(p))
#define LAS3(p) ((__attribute__((address_space(3))) v4i_*)(p))
#endif

__device__ __forceinline__ void wait_async_lds() {
#if defined(HAS_ASYNC_LDS)
#if __has_builtin(__builtin_amdgcn_s_wait_asynccnt)
    __builtin_amdgcn_s_wait_asynccnt(0);
#else
    asm volatile("s_wait_asynccnt 0x0" ::: "memory");
#endif
#endif
}

// ---------- problem geometry ----------
#define ATOM_DIM 128
#define EDGE_DIM 64
#define K_DIM    320          // 2*ATOM_DIM + EDGE_DIM
#define N_TOT    256          // [s | g]
#define KT_N     10           // K_DIM / 32
#define NT_N     16           // N_TOT / 16
#define EPW      128          // edges per workgroup
#define MROW     328          // padded merged-row stride in halfs (bank-conflict free)

#define LDS_W_HALFS (KT_N * NT_N * 512)          // 81920 halfs
#define LDS_W_BYTES (LDS_W_HALFS * 2)            // 163840 B
#define LDS_M_BYTES (EPW * MROW * 2)             // 83968 B
#define LDS_SRC_OFF (LDS_W_BYTES + LDS_M_BYTES)  // 247808 B
#define SMEM_BYTES  (LDS_SRC_OFF + EPW * 4)      // 248320 B (<320KB WGP LDS)

__device__ __forceinline__ float sigmoidf_(float x) {
    return 1.0f / (1.0f + __expf(-x));
}
__device__ __forceinline__ float softplusf_(float x) {
    return (x > 20.0f) ? x : log1pf(__expf(x));
}

// ---------------------------------------------------------------------------
// Pre-swizzle [kernel_s | kernel_g] (fp32, 320x128 each) into f16 WMMA
// B-fragment order. Fragment fid = nt*KT_N + kt holds the 32x16 (KxN) tile;
// lane l (0..31): n_local = l&15, k_local = (l<16 ? 0 : 8) + (h&7) + (h>=8 ? 16 : 0), h = 0..15,
// stored lane-contiguous so the edge kernel reads 32B per lane.
// ---------------------------------------------------------------------------
__global__ void wprep_kernel(const float* __restrict__ ks,
                             const float* __restrict__ kg,
                             _Float16* __restrict__ wsw) {
    int t = blockIdx.x * blockDim.x + threadIdx.x;
    if (t >= LDS_W_HALFS) return;
    int fid    = t >> 9;          // /512
    int within = t & 511;
    int lane   = within >> 4;
    int h      = within & 15;
    int kt = fid % KT_N;
    int nt = fid / KT_N;
    int n  = nt * 16 + (lane & 15);
    int k  = kt * 32 + ((lane < 16) ? 0 : 8) + (h & 7) + ((h & 8) << 1);
    float v = (n < ATOM_DIM) ? ks[k * ATOM_DIM + n]
                             : kg[k * ATOM_DIM + (n - ATOM_DIM)];
    wsw[t] = (_Float16)v;
}

// fp32 -> f16, 4 elements per thread
__global__ void cvt4_kernel(const float* __restrict__ in,
                            _Float16* __restrict__ out, int n4) {
    int i = blockIdx.x * blockDim.x + threadIdx.x;
    if (i >= n4) return;
    float4 v = ((const float4*)in)[i];
    H4 o = { (_Float16)v.x, (_Float16)v.y, (_Float16)v.z, (_Float16)v.w };
    ((H4*)out)[i] = o;
}

// ---------------------------------------------------------------------------
// Edge kernel: async-gather -> LDS merged tile -> WMMA GEMM (M=16/wave,
// N=256, K=320) -> sigmoid(s)*softplus(g) -> atomic scatter into agg[src].
// 256 threads = 8 waves, 128 edges per workgroup.
// ---------------------------------------------------------------------------
__global__ void __launch_bounds__(256)
edge_msg_kernel(const _Float16* __restrict__ xh,     // (n_nodes,128) f16
                const float*    __restrict__ efeat,  // (n_edges,64)  f32
                const _Float16* __restrict__ wsw,    // swizzled W    f16
                const float*    __restrict__ bs,     // bias_s (128)
                const float*    __restrict__ bg,     // bias_g (128)
                const int*      __restrict__ pair,   // (n_edges,2)
                float*          __restrict__ agg,    // (n_nodes,128) f32
                int n_edges) {
    extern __shared__ char smem[];
    _Float16* lw   = (_Float16*)smem;                    // W fragments
    _Float16* lm   = (_Float16*)(smem + LDS_W_BYTES);    // merged tile
    int*      lsrc = (int*)(smem + LDS_SRC_OFF);         // src node ids

    const int tid = threadIdx.x;

    // ---- phase 1: stage W fragments into LDS (L2-resident source) ----
    {
        const uint4* g = (const uint4*)wsw;
        uint4*       l = (uint4*)lw;
#if defined(HAS_ASYNC_LDS)
        for (int i = tid; i < LDS_W_BYTES / 16; i += 256)
            __builtin_amdgcn_global_load_async_to_lds_b128(GAS1(g + i), LAS3(l + i), 0, 0);
#else
        for (int i = tid; i < LDS_W_BYTES / 16; i += 256) l[i] = g[i];
#endif
    }

    // ---- phase 2: gather merged rows: [x[src] | x[dst] | edge_feat] ----
    {
        const int r   = tid >> 1;       // edge row in tile (2 threads/row)
        const int sel = tid & 1;        // 0 -> src half, 1 -> dst half
        const int e   = blockIdx.x * EPW + r;
        if (e < n_edges) {
            const int src = pair[2 * e];
            const int dst = pair[2 * e + 1];
            if (sel == 0) lsrc[r] = src;
            const int node = sel ? dst : src;
            const uint4* xr = (const uint4*)(xh + (size_t)node * ATOM_DIM);
            uint4* mr = (uint4*)(lm + r * MROW + sel * ATOM_DIM);
#if defined(HAS_ASYNC_LDS)
#pragma unroll
            for (int i = 0; i < 16; ++i)
                __builtin_amdgcn_global_load_async_to_lds_b128(GAS1(xr + i), LAS3(mr + i), 0, 0);
#else
#pragma unroll
            for (int i = 0; i < 16; ++i) mr[i] = xr[i];   // 128 halfs
#endif
            // edge features: fp32 -> f16 in-flight (each thread 32 floats)
            const float4* er = (const float4*)(efeat + (size_t)e * EDGE_DIM) + sel * 8;
            _Float16* md = lm + r * MROW + 2 * ATOM_DIM + sel * 32;
#pragma unroll
            for (int i = 0; i < 8; ++i) {
                float4 v = er[i];
                H4 o = { (_Float16)v.x, (_Float16)v.y, (_Float16)v.z, (_Float16)v.w };
                *(H4*)(md + i * 4) = o;
            }
        } else if (sel == 0) {
            lsrc[r] = -1;   // invalid row: computed but never scattered
        }
    }
    wait_async_lds();      // drain ASYNCcnt before the workgroup barrier
    __syncthreads();

    // ---- phase 3: WMMA GEMM + activations + scatter ----
    const int lane = tid & 31;
    const int wave = tid >> 5;
    const int m0   = wave * 16;       // this wave's 16 edge rows

    // A fragments: 16x32 f16 per ISA layout. lane<16: M=lane, K {0..7,16..23};
    // lane>=16: M=lane-16, K {8..15,24..31}  (two 16B LDS reads per fragment)
    Frag16 af[KT_N];
    {
        const int row = m0 + (lane & 15);
        const int kb  = (lane < 16) ? 0 : 8;
#pragma unroll
        for (int kt = 0; kt < KT_N; ++kt) {
            const _Float16* p = lm + row * MROW + kt * 32 + kb;
            af[kt].u[0] = *(const uint4*)(p);
            af[kt].u[1] = *(const uint4*)(p + 16);
        }
    }

    const int coln = lane & 15;       // output column within N-tile
    const int mofs = (lane < 16) ? 0 : 8;

#pragma unroll 1
    for (int ntp = 0; ntp < 8; ++ntp) {   // paired s-tile ntp / g-tile ntp+8
        const int col = ntp * 16 + coln;
        const float b0 = bs[col];
        const float b1 = bg[col];
        v8f accS, accG;
#pragma unroll
        for (int j = 0; j < 8; ++j) { accS[j] = b0; accG[j] = b1; }

#pragma unroll
        for (int kt = 0; kt < KT_N; ++kt) {
            Frag16 fbs, fbg;
            const uint4* ps = (const uint4*)(lw + (ntp * KT_N + kt) * 512 + lane * 16);
            fbs.u[0] = ps[0]; fbs.u[1] = ps[1];
            const uint4* pg = (const uint4*)(lw + ((ntp + 8) * KT_N + kt) * 512 + lane * 16);
            fbg.u[0] = pg[0]; fbg.u[1] = pg[1];
            accS = __builtin_amdgcn_wmma_f32_16x16x32_f16(
                       false, af[kt].v, false, fbs.v, (short)0, accS, false, false);
            accG = __builtin_amdgcn_wmma_f32_16x16x32_f16(
                       false, af[kt].v, false, fbg.v, (short)0, accG, false, false);
        }

        // C/D layout: lane group selects M base, VGPR j selects M row.
#pragma unroll
        for (int j = 0; j < 8; ++j) {
            const int srcn = lsrc[m0 + j + mofs];
            if (srcn >= 0) {
                const float msg = sigmoidf_(accS[j]) * softplusf_(accG[j]);
                atomicAdd(agg + (size_t)srcn * ATOM_DIM + col, msg);
            }
        }
    }
}

// x_new = softplus(x + agg); also emit f16 copy for next step's gathers
__global__ void node_update_kernel(const float* __restrict__ xin,
                                   const float* __restrict__ agg,
                                   float* __restrict__ xout,
                                   _Float16* __restrict__ xh, int n4) {
    int i = blockIdx.x * blockDim.x + threadIdx.x;
    if (i >= n4) return;
    float4 x = ((const float4*)xin)[i];
    float4 a = ((const float4*)agg)[i];
    float4 o;
    o.x = softplusf_(x.x + a.x);
    o.y = softplusf_(x.y + a.y);
    o.z = softplusf_(x.z + a.z);
    o.w = softplusf_(x.w + a.w);
    ((float4*)xout)[i] = o;
    H4 h = { (_Float16)o.x, (_Float16)o.y, (_Float16)o.z, (_Float16)o.w };
    ((H4*)xh)[i] = h;
}

// ---------------------------------------------------------------------------
extern "C" void kernel_launch(void* const* d_in, const int* in_sizes, int n_in,
                              void* d_out, int out_size, void* d_ws, size_t ws_size,
                              hipStream_t stream) {
    const float* atom  = (const float*)d_in[0];
    const float* efeat = (const float*)d_in[1];
    // d_in[2] state_attrs: unused by reference
    const float* ks = (const float*)d_in[3];
    const float* bs = (const float*)d_in[4];
    const float* kg = (const float*)d_in[5];
    const float* bg = (const float*)d_in[6];
    const int*   pr = (const int*)d_in[7];
    // d_in[8], d_in[9]: unused by reference

    const int n_nodes = in_sizes[0] / ATOM_DIM;
    const int n_edges = in_sizes[7] / 2;
    float* out = (float*)d_out;

    // workspace carve-up (all 256B aligned)
    char* ws = (char*)d_ws;
    _Float16* wsw = (_Float16*)ws;  ws += LDS_W_BYTES;                    // 160 KB
    _Float16* xh  = (_Float16*)ws;  ws += (size_t)n_nodes * ATOM_DIM * 2; // 12.8 MB
    float* agg    = (float*)ws;     ws += (size_t)n_nodes * ATOM_DIM * 4; // 25.6 MB
    float* xbuf   = (float*)ws;                                           // 25.6 MB

    // allow >64KB dynamic LDS (idempotent, capture-safe host call)
    (void)hipFuncSetAttribute(reinterpret_cast<const void*>(edge_msg_kernel),
                              hipFuncAttributeMaxDynamicSharedMemorySize,
                              SMEM_BYTES);

    // one-time (per call) preps
    wprep_kernel<<<(LDS_W_HALFS + 255) / 256, 256, 0, stream>>>(ks, kg, wsw);
    const int n4x = n_nodes * ATOM_DIM / 4;
    cvt4_kernel<<<(n4x + 255) / 256, 256, 0, stream>>>(atom, xh, n4x);

    const int nblk = (n_edges + EPW - 1) / EPW;
    for (int s = 0; s < 3; ++s) {
        (void)hipMemsetAsync(agg, 0, (size_t)n_nodes * ATOM_DIM * 4, stream);
        edge_msg_kernel<<<nblk, 256, SMEM_BYTES, stream>>>(
            xh, efeat, wsw, bs, bg, pr, agg, n_edges);
        const float* xi = (s == 0) ? atom : xbuf;
        float*       xo = (s == 2) ? out  : xbuf;
        node_update_kernel<<<(n4x + 255) / 256, 256, 0, stream>>>(xi, agg, xo, xh, n4x);
    }
}
